// SatGatewayGNN_70978629533813
// MI455X (gfx1250) — matrix operand order repaired and checked by reading, not compiled, not run on hardware
//
#include <hip/hip_runtime.h>
#include <cstddef>
#include <cstdint>

// ---------------- problem constants ----------------
#define NS   8192
#define NC   1024
#define NG   64
#define HD   512
#define GD   64
#define CDIM 1024
#define IND  1600   // HD + GD + CDIM

typedef __bf16 bf16;
typedef __attribute__((ext_vector_type(16))) __bf16 v16bf;
typedef __attribute__((ext_vector_type(4)))  __bf16 v4bf;
typedef __attribute__((ext_vector_type(8)))  float  v8f;

#define TM  128
#define TN  64
#define TK  32
#define TKP 40   // padded LDS row (80B = 16B multiple)

// ---- CDNA5 async copy: global -> LDS, ASYNCcnt-tracked (ISA §15.18.3) ----
__device__ __forceinline__ void async_b128(void* lds, const void* g) {
  unsigned l = (unsigned)(uintptr_t)lds;   // generic addr low 32 bits == LDS offset
  asm volatile("global_load_async_to_lds_b128 %0, %1, off"
               :: "v"(l), "v"(g) : "memory");
}

__device__ __forceinline__ void compute_tile(bf16 (*__restrict__ Asb)[TKP],
                                             bf16 (*__restrict__ Wsb)[TKP],
                                             int wm, int wn, int lane,
                                             v8f acc[2][2]) {
  const int row = lane & 15;
  const int hi  = lane >> 4;
  v16bf afr[2], bfr[2];
#pragma unroll
  for (int t = 0; t < 2; ++t) {
    const bf16* ap = &Asb[wm + t * 16 + row][0];
    const bf16* bp = &Wsb[wn + t * 16 + row][0];
#pragma unroll
    for (int e = 0; e < 8; ++e) {
      afr[t][e]     = ap[hi * 8 + e];        // A 16x32: K {0..7|8..15} (+16)
      afr[t][e + 8] = ap[hi * 8 + 16 + e];
      bfr[t][e]     = bp[hi * 8 + e];        // B 32x16: K {0..7|8..15} (+16)
      bfr[t][e + 8] = bp[hi * 8 + 16 + e];
    }
  }
#pragma unroll
  for (int i = 0; i < 2; ++i)
#pragma unroll
    for (int j = 0; j < 2; ++j)
      acc[i][j] = __builtin_amdgcn_wmma_f32_16x16x32_bf16(
          false, afr[i], false, bfr[j], (short)0, acc[i][j], false, false);
}

// =====================================================================
// WMMA GEMM:  C[M,N] (op: = or +=)  A[M,K]bf16 @ W[N,K]^T bf16 (+ bias[N])
// M mult of 128, N mult of 64, K mult of 16. 256 threads = 8 wave32,
// 128x64 tile, triple-buffered LDS fed by async b128 copies; each wave
// owns a 32x32 quadrant = 4 v_wmma_f32_16x16x32_bf16 per k-step.
// =====================================================================
__global__ __launch_bounds__(256)
void gemm_wmma_bf16(const bf16* __restrict__ A, const bf16* __restrict__ W,
                    const float* __restrict__ bias, float* __restrict__ C,
                    int M, int N, int K, int accumulate) {
  __shared__ bf16 As[3][TM][TKP];
  __shared__ bf16 Ws[3][TN][TKP];

  const int mBase = blockIdx.y * TM;
  const int nBase = blockIdx.x * TN;
  const int tid   = threadIdx.x;
  const int lane  = tid & 31;
  const int wave  = tid >> 5;
  const int wm    = (wave & 3) * 32;
  const int wn    = (wave >> 2) * 32;

  // staging coords: 8-bf16 (16B) chunks. A: rows {tid>>2, +64}; W: row tid>>2
  const int ar = tid >> 2;
  const int ac = (tid & 3) << 3;

  const int nk = K / TK;
  v8f acc[2][2] = {};

  if (nk > 0) {
    // prologue: tile 0 -> buf 0
    async_b128(&As[0][ar][ac],      A + (size_t)(mBase + ar) * K + ac);
    async_b128(&As[0][ar + 64][ac], A + (size_t)(mBase + ar + 64) * K + ac);
    async_b128(&Ws[0][ar][ac],      W + (size_t)(nBase + ar) * K + ac);
    for (int t = 0; t < nk; ++t) {
      const int buf = t % 3;
      if (t + 1 < nk) {
        const int nb = (t + 1) % 3;
        const size_t ko = (size_t)(t + 1) * TK;
        async_b128(&As[nb][ar][ac],      A + (size_t)(mBase + ar) * K + ko + ac);
        async_b128(&As[nb][ar + 64][ac], A + (size_t)(mBase + ar + 64) * K + ko + ac);
        async_b128(&Ws[nb][ar][ac],      W + (size_t)(nBase + ar) * K + ko + ac);
        asm volatile("s_wait_asynccnt 0x3" ::: "memory");  // tile t resident
      } else {
        asm volatile("s_wait_asynccnt 0x0" ::: "memory");
      }
      __syncthreads();
      compute_tile(As[buf], Ws[buf], wm, wn, lane, acc);
    }
  }
  if (nk * TK < K) {            // K tail (K=16 init GEMMs only)
    const int k0 = nk * TK;
    __syncthreads();
    bf16 t0[8], t1[8], tw[8];
#pragma unroll
    for (int e = 0; e < 8; ++e) {
      int gk = k0 + ac + e;
      t0[e] = (gk < K) ? A[(size_t)(mBase + ar) * K + gk] : (bf16)0.0f;
      t1[e] = (gk < K) ? A[(size_t)(mBase + ar + 64) * K + gk] : (bf16)0.0f;
      tw[e] = (gk < K) ? W[(size_t)(nBase + ar) * K + gk] : (bf16)0.0f;
    }
#pragma unroll
    for (int e = 0; e < 8; ++e) {
      As[0][ar][ac + e] = t0[e];
      As[0][ar + 64][ac + e] = t1[e];
      Ws[0][ar][ac + e] = tw[e];
    }
    __syncthreads();
    compute_tile(As[0], Ws[0], wm, wn, lane, acc);
  }

  // ---- epilogue: f32 C/D layout (lane = N, vgpr(+8) = M) ----
  const int nl   = lane & 15;
  const int hsel = (lane >> 4) * 8;
#pragma unroll
  for (int i = 0; i < 2; ++i)
#pragma unroll
    for (int j = 0; j < 2; ++j) {
      int ncol = nBase + wn + j * 16 + nl;
      float bv = bias ? bias[ncol] : 0.0f;
#pragma unroll
      for (int r = 0; r < 8; ++r) {
        int mrow = mBase + wm + i * 16 + r + hsel;
        size_t idx = (size_t)mrow * N + ncol;
        float v = acc[i][j][r] + bv;
        if (accumulate) C[idx] += v; else C[idx] = v;
      }
    }
}

// =====================================================================
// dtype conversion + segment-mean machinery
// =====================================================================
__global__ void f32_to_bf16_k(const float* __restrict__ in,
                              bf16* __restrict__ out, size_t n4) {
  for (size_t i = (size_t)blockIdx.x * blockDim.x + threadIdx.x; i < n4;
       i += (size_t)gridDim.x * blockDim.x) {
    float4 v = ((const float4*)in)[i];
    v4bf p;
    p[0] = (bf16)v.x; p[1] = (bf16)v.y; p[2] = (bf16)v.z; p[3] = (bf16)v.w;
    ((v4bf*)out)[i] = p;
  }
}

__global__ void fill_zero_k(float* __restrict__ p, size_t n) {
  for (size_t i = (size_t)blockIdx.x * blockDim.x + threadIdx.x; i < n;
       i += (size_t)gridDim.x * blockDim.x)
    p[i] = 0.0f;
}

__global__ void deg_count_k(const int* __restrict__ dst, int E,
                            float* __restrict__ deg) {
  int i = blockIdx.x * blockDim.x + threadIdx.x;
  if (i < E) atomicAdd(&deg[dst[i]], 1.0f);
}

// single-block exclusive scan of (int)deg -> offs and cursor copies
__global__ __launch_bounds__(1024)
void exscan_k(const float* __restrict__ deg, int n, int* __restrict__ offs,
              int* __restrict__ cursor) {
  __shared__ int part[1024];
  const int tid = threadIdx.x;
  const int per = (n + 1023) >> 10;
  const int base = tid * per;
  int s = 0;
  for (int i = 0; i < per; ++i) {
    int idx = base + i;
    if (idx < n) s += (int)deg[idx];
  }
  part[tid] = s;
  __syncthreads();
  for (int off = 1; off < 1024; off <<= 1) {
    int v = (tid >= off) ? part[tid - off] : 0;
    __syncthreads();
    if (tid >= off) part[tid] += v;
    __syncthreads();
  }
  int run = (tid > 0) ? part[tid - 1] : 0;
  for (int i = 0; i < per; ++i) {
    int idx = base + i;
    if (idx >= n) break;
    offs[idx] = run; cursor[idx] = run;
    run += (int)deg[idx];
  }
}

__global__ void csr_fill_k(const int* __restrict__ src, const int* __restrict__ dst,
                           int E, int* __restrict__ cursor, int* __restrict__ csr_src) {
  int i = blockIdx.x * blockDim.x + threadIdx.x;
  if (i < E) {
    int p = atomicAdd(&cursor[dst[i]], 1);
    csr_src[p] = src[i];
  }
}

// out[n,f] (=|+=) mean_{e in N(n)} h[csr_src[e], f] + bias[f]; 1 block/row
__global__ __launch_bounds__(256)
void gather_mean_k(const float* __restrict__ h, const int* __restrict__ offs,
                   const float* __restrict__ deg, const int* __restrict__ csr_src,
                   const float* __restrict__ bias, float* __restrict__ out,
                   int F, int accumulate) {
  const int n = blockIdx.x;
  const int start = offs[n];
  const int cnt = (int)deg[n];
  const float inv = 1.0f / (float)(cnt > 1 ? cnt : 1);
  for (int f = threadIdx.x; f < F; f += 256) {
    float s = 0.0f;
    for (int e = 0; e < cnt; ++e)
      s += h[(size_t)csr_src[start + e] * F + f];
    float v = s * inv + (bias ? bias[f] : 0.0f);
    size_t idx = (size_t)n * F + f;
    if (accumulate) out[idx] += v; else out[idx] = v;
  }
}

// atomic scatter (init convs only, F<=32)
__global__ void scatter_add_k(const float* __restrict__ feat,
                              const int* __restrict__ src,
                              const int* __restrict__ dst,
                              int E, int F, float* __restrict__ msg) {
  size_t total = (size_t)E * (size_t)F;
  for (size_t i = (size_t)blockIdx.x * blockDim.x + threadIdx.x; i < total;
       i += (size_t)gridDim.x * blockDim.x) {
    size_t e = i / (size_t)F;
    int f = (int)(i - e * (size_t)F);
    atomicAdd(&msg[(size_t)dst[e] * F + f], feat[(size_t)src[e] * F + f]);
  }
}

// mean(raw) -> bf16 (feeds GEMM A side)
__global__ void finalize_agg_k(const float* __restrict__ msg,
                               const float* __restrict__ deg,
                               bf16* __restrict__ out, int Nn, int F) {
  size_t total = (size_t)Nn * (size_t)F;
  for (size_t i = (size_t)blockIdx.x * blockDim.x + threadIdx.x; i < total;
       i += (size_t)gridDim.x * blockDim.x) {
    size_t n = i / (size_t)F;
    float d = deg[n]; d = d > 1.0f ? d : 1.0f;
    out[i] = (bf16)(msg[i] / d);
  }
}

// relu, f32 acc -> bf16 activation
__global__ void relu_bf16_k(const float* __restrict__ in, bf16* __restrict__ out,
                            size_t n4) {
  for (size_t i = (size_t)blockIdx.x * blockDim.x + threadIdx.x; i < n4;
       i += (size_t)gridDim.x * blockDim.x) {
    float4 v = ((const float4*)in)[i];
    v4bf p;
    p[0] = (bf16)fmaxf(v.x, 0.0f); p[1] = (bf16)fmaxf(v.y, 0.0f);
    p[2] = (bf16)fmaxf(v.z, 0.0f); p[3] = (bf16)fmaxf(v.w, 0.0f);
    ((v4bf*)out)[i] = p;
  }
}

// sat_in(bf16) = concat([sat bf16 (512), gwl f32 (64), cl f32 (1024)])
__global__ void concat_satin_k(const bf16* __restrict__ sat,
                               const float* __restrict__ gwl,
                               const float* __restrict__ cl,
                               bf16* __restrict__ satin) {
  size_t total = (size_t)NS * IND;
  for (size_t i = (size_t)blockIdx.x * blockDim.x + threadIdx.x; i < total;
       i += (size_t)gridDim.x * blockDim.x) {
    size_t s = i / IND;
    int j = (int)(i - s * IND);
    bf16 v;
    if (j < HD)            v = sat[s * HD + j];
    else if (j < HD + GD)  v = (bf16)gwl[s * GD + (j - HD)];
    else                   v = (bf16)cl[s * CDIM + (j - HD - GD)];
    satin[i] = v;
  }
}

// dem[s] = sum_c sigmoid(cl[s,c]) * cvm[s,c] * demands[c]   (one wave32/row)
__global__ __launch_bounds__(256)
void demand_k(const float* __restrict__ cl, const float* __restrict__ cvm,
              const float* __restrict__ demands, float* __restrict__ dem) {
  int row  = blockIdx.x * 8 + (threadIdx.x >> 5);
  int lane = threadIdx.x & 31;
  if (row >= NS) return;
  float s = 0.0f;
  for (int c = lane; c < CDIM; c += 32) {
    float x  = cl[(size_t)row * CDIM + c];
    float sg = 1.0f / (1.0f + expf(-x));
    s += sg * cvm[(size_t)row * CDIM + c] * demands[c];
  }
#pragma unroll
  for (int o = 16; o > 0; o >>= 1) s += __shfl_down(s, o, 32);
  if (lane == 0) dem[row] = s;
}

// =====================================================================
// host-side launch helpers
// =====================================================================
static inline void gemm(const bf16* A, const bf16* W, const float* b, float* C,
                        int M, int N, int K, int acc, hipStream_t st) {
  dim3 g(N / TN, M / TM);
  gemm_wmma_bf16<<<g, 256, 0, st>>>(A, W, b, C, M, N, K, acc);
}
static inline void zero(float* p, size_t n, hipStream_t st) {
  int blocks = (int)((n + 255) / 256); if (blocks > 16384) blocks = 16384;
  fill_zero_k<<<blocks, 256, 0, st>>>(p, n);
}
static inline void scatter(const float* f, const int* s, const int* d, int E,
                           int F, float* msg, hipStream_t st) {
  size_t total = (size_t)E * F;
  int blocks = (int)((total + 255) / 256); if (blocks > 65536) blocks = 65536;
  scatter_add_k<<<blocks, 256, 0, st>>>(f, s, d, E, F, msg);
}
static inline void relu_bf(const float* in, bf16* out, size_t n, hipStream_t st) {
  size_t n4 = n / 4;
  int blocks = (int)((n4 + 255) / 256); if (blocks > 16384) blocks = 16384;
  relu_bf16_k<<<blocks, 256, 0, st>>>(in, out, n4);
}

extern "C" void kernel_launch(void* const* d_in, const int* in_sizes, int n_in,
                              void* d_out, int out_size, void* d_ws, size_t ws_size,
                              hipStream_t stream) {
  // ---- inputs (setup_inputs insertion order; params in pytree order) ----
  const float* x_sat  = (const float*)d_in[0];
  const float* x_gw   = (const float*)d_in[1];
  const float* x_cell = (const float*)d_in[2];
  const int* e_ss = (const int*)d_in[3]; int ESS = in_sizes[3] / 2;
  /* e_sg (d_in[4]) unused: init gateway conv output never read */
  const int* e_gc = (const int*)d_in[5]; int EGC = in_sizes[5] / 2;
  const int* e_sc = (const int*)d_in[6]; int ESC = in_sizes[6] / 2;
  const int* e_cs = (const int*)d_in[7]; int ECS = in_sizes[7] / 2;
  const int *ss_s = e_ss, *ss_d = e_ss + ESS;
  const int *gc_s = e_gc, *gc_d = e_gc + EGC;
  const int *sc_s = e_sc, *sc_d = e_sc + ESC;
  const int *cs_s = e_cs, *cs_d = e_cs + ECS;

  // params: cell_heads(8), gw_heads(8), init{cs,gc,sc,sg,ss}(15), rounds 3x{cs,sc,ss}(27)
  const float *cellW[4], *cellB[4], *gwW[4], *gwB[4];
  for (int i = 0; i < 4; ++i) {
    cellW[i] = (const float*)d_in[8 + 2 * i];
    cellB[i] = (const float*)d_in[9 + 2 * i];
    gwW[i]   = (const float*)d_in[16 + 2 * i];
    gwB[i]   = (const float*)d_in[17 + 2 * i];
  }
  const float* iCsWl = (const float*)d_in[24]; const float* iCsBl = (const float*)d_in[25];
  const float* iCsWr = (const float*)d_in[26];
  const float* iGcWl = (const float*)d_in[27]; const float* iGcBl = (const float*)d_in[28];
  const float* iGcWr = (const float*)d_in[29];
  const float* iScWl = (const float*)d_in[30]; const float* iScBl = (const float*)d_in[31];
  const float* iScWr = (const float*)d_in[32];
  const float* iSsWl = (const float*)d_in[36]; const float* iSsBl = (const float*)d_in[37];
  const float* iSsWr = (const float*)d_in[38];
  const float* rCsWl[3], *rCsBl[3], *rCsWr[3], *rScWl[3], *rScBl[3], *rScWr[3],
             * rSsWl[3], *rSsBl[3], *rSsWr[3];
  for (int r = 0; r < 3; ++r) {
    int b = 39 + 9 * r;
    rCsWl[r] = (const float*)d_in[b + 0]; rCsBl[r] = (const float*)d_in[b + 1];
    rCsWr[r] = (const float*)d_in[b + 2];
    rScWl[r] = (const float*)d_in[b + 3]; rScBl[r] = (const float*)d_in[b + 4];
    rScWr[r] = (const float*)d_in[b + 5];
    rSsWl[r] = (const float*)d_in[b + 6]; rSsBl[r] = (const float*)d_in[b + 7];
    rSsWr[r] = (const float*)d_in[b + 8];
  }
  const float* cvm     = (const float*)d_in[66];
  const float* demands = (const float*)d_in[67];

  // ---- outputs: [4,NS,GD] gw, [4,NS,CDIM] cell, [4,NS] dem ----
  float* out_gw   = (float*)d_out;
  float* out_cell = out_gw + (size_t)4 * NS * GD;
  float* out_dem  = out_cell + (size_t)4 * NS * CDIM;

  // ---- workspace layout: f32 region | int region | bf16 region ----
  float* f = (float*)d_ws;
  float* b_h    = f;  f += (size_t)NS * HD;   // GEMM scratch (f32 C output)
  float* b_msg  = f;  f += (size_t)NS * 32;   // init raw-scatter only
  float* b_accS = f;  f += (size_t)NS * HD;
  float* b_accC = f;  f += (size_t)NC * HD;
  float* deg_ss = f;  f += NS;
  float* deg_cs = f;  f += NS;
  float* deg_sc = f;  f += NC;
  float* deg_gc = f;  f += NC;
  int* iw = (int*)f;
  int* offs_ss = iw;  iw += NS;
  int* cur_ss  = iw;  iw += NS;
  int* offs_cs = iw;  iw += NS;
  int* cur_cs  = iw;  iw += NS;
  int* offs_sc = iw;  iw += NC;
  int* cur_sc  = iw;  iw += NC;
  int* csr_ss  = iw;  iw += ESS;
  int* csr_cs  = iw;  iw += ECS;
  int* csr_sc  = iw;  iw += ESC;
  bf16* bb = (bf16*)iw;
  bf16* b_sat   = bb;  bb += (size_t)NS * HD;
  bf16* b_cell  = bb;  bb += (size_t)NC * HD;
  bf16* b_satin = bb;  bb += (size_t)NS * IND;
  bf16* b_hb    = bb;  bb += (size_t)NS * 32;   // bf16 raw-mean (init)
  bf16* arena   = bb;                            // bf16 weight/input copies

  // one-pass fp32 -> bf16 conversion (bump allocator; fixed order per call)
  auto cv = [&](const float* src, size_t n) -> bf16* {
    bf16* dst = arena; arena += n;
    size_t n4 = n / 4;
    int blocks = (int)((n4 + 255) / 256); if (blocks > 8192) blocks = 8192;
    f32_to_bf16_k<<<blocks, 256, 0, stream>>>(src, dst, n4);
    return dst;
  };
  bf16* bx_sat  = cv(x_sat,  (size_t)NS * 32);
  bf16* bx_cell = cv(x_cell, (size_t)NC * 16);
  bf16 *bcellW[4], *bgwW[4];
  for (int i = 0; i < 4; ++i) {
    bcellW[i] = cv(cellW[i], (size_t)CDIM * HD);
    bgwW[i]   = cv(gwW[i],   (size_t)GD * HD);
  }
  bf16* biSsWl = cv(iSsWl, (size_t)HD * 32);
  bf16* biSsWr = cv(iSsWr, (size_t)HD * 32);
  bf16* biCsWl = cv(iCsWl, (size_t)HD * 16);
  bf16* biCsWr = cv(iCsWr, (size_t)HD * 32);
  bf16* biGcWl = cv(iGcWl, (size_t)HD * 16);
  bf16* biGcWr = cv(iGcWr, (size_t)HD * 16);
  bf16* biScWl = cv(iScWl, (size_t)HD * 32);
  bf16* biScWr = cv(iScWr, (size_t)HD * 16);
  bf16 *brCsWl[3], *brCsWr[3], *brScWl[3], *brScWr[3], *brSsWl[3], *brSsWr[3];
  for (int r = 0; r < 3; ++r) {
    brCsWl[r] = cv(rCsWl[r], (size_t)HD * HD);
    brCsWr[r] = cv(rCsWr[r], (size_t)HD * IND);
    brScWl[r] = cv(rScWl[r], (size_t)HD * IND);
    brScWr[r] = cv(rScWr[r], (size_t)HD * HD);
    brSsWl[r] = cv(rSsWl[r], (size_t)HD * IND);
    brSsWr[r] = cv(rSsWr[r], (size_t)HD * IND);
  }

  // ---- degrees + CSR (reused across all rounds) ----
  zero(deg_ss, (size_t)NS + NS + NC + NC, stream);
  deg_count_k<<<(ESS + 255) / 256, 256, 0, stream>>>(ss_d, ESS, deg_ss);
  deg_count_k<<<(ECS + 255) / 256, 256, 0, stream>>>(cs_d, ECS, deg_cs);
  deg_count_k<<<(ESC + 255) / 256, 256, 0, stream>>>(sc_d, ESC, deg_sc);
  deg_count_k<<<(EGC + 255) / 256, 256, 0, stream>>>(gc_d, EGC, deg_gc);
  exscan_k<<<1, 1024, 0, stream>>>(deg_ss, NS, offs_ss, cur_ss);
  exscan_k<<<1, 1024, 0, stream>>>(deg_cs, NS, offs_cs, cur_cs);
  exscan_k<<<1, 1024, 0, stream>>>(deg_sc, NC, offs_sc, cur_sc);
  csr_fill_k<<<(ESS + 255) / 256, 256, 0, stream>>>(ss_s, ss_d, ESS, cur_ss, csr_ss);
  csr_fill_k<<<(ECS + 255) / 256, 256, 0, stream>>>(cs_s, cs_d, ECS, cur_cs, csr_cs);
  csr_fill_k<<<(ESC + 255) / 256, 256, 0, stream>>>(sc_s, sc_d, ESC, cur_sc, csr_sc);

  // ==================== init conv: sat ====================
  zero(b_msg, (size_t)NS * 32, stream);
  scatter(x_sat, ss_s, ss_d, ESS, 32, b_msg, stream);
  finalize_agg_k<<<1024, 256, 0, stream>>>(b_msg, deg_ss, b_hb, NS, 32);
  gemm(b_hb,   biSsWl, iSsBl,  b_accS, NS, HD, 32, 0, stream);
  gemm(bx_sat, biSsWr, nullptr, b_accS, NS, HD, 32, 1, stream);
  zero(b_msg, (size_t)NS * 16, stream);
  scatter(x_cell, cs_s, cs_d, ECS, 16, b_msg, stream);
  finalize_agg_k<<<512, 256, 0, stream>>>(b_msg, deg_cs, b_hb, NS, 16);
  gemm(b_hb,   biCsWl, iCsBl,  b_accS, NS, HD, 16, 1, stream);
  gemm(bx_sat, biCsWr, nullptr, b_accS, NS, HD, 32, 1, stream);
  relu_bf(b_accS, b_sat, (size_t)NS * HD, stream);

  // ==================== init conv: cell ====================
  zero(b_msg, (size_t)NC * 16, stream);
  scatter(x_gw, gc_s, gc_d, EGC, 16, b_msg, stream);
  finalize_agg_k<<<64, 256, 0, stream>>>(b_msg, deg_gc, b_hb, NC, 16);
  gemm(b_hb,    biGcWl, iGcBl,  b_accC, NC, HD, 16, 0, stream);
  gemm(bx_cell, biGcWr, nullptr, b_accC, NC, HD, 16, 1, stream);
  zero(b_msg, (size_t)NC * 32, stream);
  scatter(x_sat, sc_s, sc_d, ESC, 32, b_msg, stream);
  finalize_agg_k<<<128, 256, 0, stream>>>(b_msg, deg_sc, b_hb, NC, 32);
  gemm(b_hb,    biScWl, iScBl,  b_accC, NC, HD, 32, 1, stream);
  gemm(bx_cell, biScWr, nullptr, b_accC, NC, HD, 16, 1, stream);
  relu_bf(b_accC, b_cell, (size_t)NC * HD, stream);

  // ==================== heads(0) + sat_in ====================
  {
    float* gwl = out_gw;  float* cl = out_cell;  float* dm = out_dem;
    gemm(b_sat, bgwW[0],   gwB[0],   gwl, NS, GD,   HD, 0, stream);
    gemm(b_sat, bcellW[0], cellB[0], cl,  NS, CDIM, HD, 0, stream);
    demand_k<<<NS / 8, 256, 0, stream>>>(cl, cvm, demands, dm);
    concat_satin_k<<<16384, 256, 0, stream>>>(b_sat, gwl, cl, b_satin);
  }

  // ==================== message-passing rounds ====================
  for (int r = 0; r < 3; ++r) {
    // ---- cell accumulator (uses OLD cell as x_dst) ----
    gemm(b_satin, brScWl[r], nullptr, b_h, NS, HD, IND, 0, stream);  // sat_in @ Wl_sc
    gather_mean_k<<<NC, 256, 0, stream>>>(b_h, offs_sc, deg_sc, csr_sc,
                                          rScBl[r], b_accC, HD, 0);
    gemm(b_cell, brScWr[r], nullptr, b_accC, NC, HD, HD, 1, stream);

    // ---- sat accumulator: ss then cs ----
    gemm(b_satin, brSsWl[r], nullptr, b_h, NS, HD, IND, 0, stream);  // sat_in @ Wl_ss
    gather_mean_k<<<NS, 256, 0, stream>>>(b_h, offs_ss, deg_ss, csr_ss,
                                          rSsBl[r], b_accS, HD, 0);
    gemm(b_satin, brSsWr[r], nullptr, b_accS, NS, HD, IND, 1, stream);

    gemm(b_cell, brCsWl[r], nullptr, b_h, NC, HD, HD, 0, stream);    // cell @ Wl_cs
    gather_mean_k<<<NS, 256, 0, stream>>>(b_h, offs_cs, deg_cs, csr_cs,
                                          rCsBl[r], b_accS, HD, 1);
    gemm(b_satin, brCsWr[r], nullptr, b_accS, NS, HD, IND, 1, stream);

    // ---- activations ----
    relu_bf(b_accS, b_sat, (size_t)NS * HD, stream);
    relu_bf(b_accC, b_cell, (size_t)NC * HD, stream);

    // ---- heads(r+1) + next sat_in ----
    float* gwl = out_gw   + (size_t)(r + 1) * NS * GD;
    float* cl  = out_cell + (size_t)(r + 1) * NS * CDIM;
    float* dm  = out_dem  + (size_t)(r + 1) * NS;
    gemm(b_sat, bgwW[r + 1],   gwB[r + 1],   gwl, NS, GD,   HD, 0, stream);
    gemm(b_sat, bcellW[r + 1], cellB[r + 1], cl,  NS, CDIM, HD, 0, stream);
    demand_k<<<NS / 8, 256, 0, stream>>>(cl, cvm, demands, dm);
    if (r < 2) {
      concat_satin_k<<<16384, 256, 0, stream>>>(b_sat, gwl, cl, b_satin);
    }
  }
}
